// Net_32323923870241
// MI455X (gfx1250) — matrix operand order repaired and checked by reading, not compiled, not run on hardware
//
#include <hip/hip_runtime.h>
#include <hip/hip_bf16.h>

#define N_NODES 250000
#define DIM     128
#define NSCALES 6
#define NEDGE   500000
#define NEDGE_LR 50000
#define NITERS  4

// LDS row stride for the staged 128x128 bf16 weight matrix, in dwords.
// 64 data dwords + 4 pad; multiple of 4 so every 16B chunk is 16B-aligned
// (required for global_load_async_to_lds_b128 LDS writes).
#define WSTRIDE 68

typedef __attribute__((ext_vector_type(16))) __bf16 v16bf;
typedef __attribute__((ext_vector_type(8)))  float  v8f;

struct alignas(16) U128 { unsigned long long lo, hi; };

union Frag16 {
    v16bf bf;
    unsigned long long d[4];
    U128 q[2];
};

__device__ __forceinline__ unsigned short f2bf(float f) {
    unsigned int u = __float_as_uint(f);
    u += 0x7fffu + ((u >> 16) & 1u);   // round-to-nearest-even
    return (unsigned short)(u >> 16);
}

// ---------------------------------------------------------------------------
// Stage a 128x128 bf16 weight matrix (32 KB) from global into LDS using the
// gfx1250 async global->LDS DMA path (ASYNCcnt). Each of 256 threads issues
// 8 x 16B transfers; LDS rows are padded to WSTRIDE dwords.
// ---------------------------------------------------------------------------
__device__ __forceinline__ void stage_weights_async(
    const unsigned short* __restrict__ W, unsigned int* lds_w)
{
#pragma unroll
    for (int c = 0; c < 8; ++c) {
        const int t   = threadIdx.x + c * 256;   // 2048 x 16B = 32 KB
        const int row = t >> 4;                  // 16 chunks per 64-dword row
        const int q   = t & 15;
        const unsigned int ldsAddr =
            (unsigned int)(size_t)(&lds_w[row * WSTRIDE + q * 4]);
        const unsigned int goff = (unsigned int)t * 16u;
        asm volatile("global_load_async_to_lds_b128 %0, %1, %2"
                     :: "v"(ldsAddr), "v"(goff), "s"(W)
                     : "memory");
    }
    asm volatile("s_wait_asynccnt 0" ::: "memory");
    __syncthreads();
}

// ---------------------------------------------------------------------------
// Dense row GEMM: Out[16t+m, :] = bf16(X[16t+m, :]) @ W.T   (W staged in LDS)
// One wave = 16 rows x 128 cols, 4 K-chunks of 32, 8 N-tiles of 16.
// ---------------------------------------------------------------------------
__global__ void __launch_bounds__(256) gemm_rows_kernel(
    const unsigned short* __restrict__ X,      // [N,128] bf16 row-major
    const unsigned short* __restrict__ W,      // [128,128] bf16 row-major
    float* __restrict__ Out,                   // [N,128] f32
    int numTiles)
{
    __shared__ unsigned int lds_w[128 * WSTRIDE];
    stage_weights_async(W, lds_w);

    const int lane = threadIdx.x & 31;
    const int tile = blockIdx.x * 8 + (threadIdx.x >> 5);
    if (tile >= numTiles) return;
    const int m    = lane & 15;
    const int half = lane >> 4;
    const int rowA = tile * 16 + m;

    const v8f vzero = {0.f, 0.f, 0.f, 0.f, 0.f, 0.f, 0.f, 0.f};
    v8f acc[8];
#pragma unroll
    for (int nt = 0; nt < 8; ++nt) acc[nt] = vzero;

    const U128* Xv = reinterpret_cast<const U128*>(X);   // 16 U128 per row
    const unsigned long long* Lq = reinterpret_cast<const unsigned long long*>(lds_w);

#pragma unroll
    for (int kc = 0; kc < 4; ++kc) {
        Frag16 a;
        // A 16x32 bf16 layout: lane(half=0): K 0..7 and 16..23; half=1: +8
        a.q[0] = Xv[rowA * 16 + kc * 4 + half];
        a.q[1] = Xv[rowA * 16 + kc * 4 + half + 2];
#pragma unroll
        for (int nt = 0; nt < 8; ++nt) {
            Frag16 b;
            const int n  = nt * 16 + m;                            // B col = W row
            const int o2 = (n * WSTRIDE + kc * 16 + half * 4) >> 1; // qword index
            b.d[0] = Lq[o2 + 0];
            b.d[1] = Lq[o2 + 1];
            b.d[2] = Lq[o2 + 4];
            b.d[3] = Lq[o2 + 5];
            acc[nt] = __builtin_amdgcn_wmma_f32_16x16x32_bf16(
                false, a.bf, false, b.bf, (short)0, acc[nt], false, false);
        }
    }

    float* outRow = Out + (size_t)(tile * 16) * DIM;
#pragma unroll
    for (int nt = 0; nt < 8; ++nt)
#pragma unroll
        for (int r = 0; r < 8; ++r)
            outRow[(size_t)(r + 8 * half) * DIM + nt * 16 + m] = acc[nt][r];
}

// ---------------------------------------------------------------------------
// Edge GEMM with gather/scatter: Out[u[e], :] += bf16(X[v[e], :]) @ W.T
// ---------------------------------------------------------------------------
__global__ void __launch_bounds__(256) gemm_edges_kernel(
    const unsigned short* __restrict__ X,
    const unsigned short* __restrict__ W,
    float* __restrict__ Out,
    const int* __restrict__ uIdx,
    const int* __restrict__ vIdx,
    int numTiles)
{
    __shared__ unsigned int lds_w[128 * WSTRIDE];
    stage_weights_async(W, lds_w);

    const int lane = threadIdx.x & 31;
    const int tile = blockIdx.x * 8 + (threadIdx.x >> 5);
    if (tile >= numTiles) return;
    const int m    = lane & 15;
    const int half = lane >> 4;

    const int rowA = vIdx[tile * 16 + m];      // gathered source row
    int urow[8];                               // scatter rows for this lane's D elems
#pragma unroll
    for (int r = 0; r < 8; ++r) urow[r] = uIdx[tile * 16 + r + 8 * half];

    const v8f vzero = {0.f, 0.f, 0.f, 0.f, 0.f, 0.f, 0.f, 0.f};
    v8f acc[8];
#pragma unroll
    for (int nt = 0; nt < 8; ++nt) acc[nt] = vzero;

    const U128* Xv = reinterpret_cast<const U128*>(X);
    const unsigned long long* Lq = reinterpret_cast<const unsigned long long*>(lds_w);

#pragma unroll
    for (int kc = 0; kc < 4; ++kc) {
        Frag16 a;
        a.q[0] = Xv[(size_t)rowA * 16 + kc * 4 + half];
        a.q[1] = Xv[(size_t)rowA * 16 + kc * 4 + half + 2];
#pragma unroll
        for (int nt = 0; nt < 8; ++nt) {
            Frag16 b;
            const int n  = nt * 16 + m;
            const int o2 = (n * WSTRIDE + kc * 16 + half * 4) >> 1;
            b.d[0] = Lq[o2 + 0];
            b.d[1] = Lq[o2 + 1];
            b.d[2] = Lq[o2 + 4];
            b.d[3] = Lq[o2 + 5];
            acc[nt] = __builtin_amdgcn_wmma_f32_16x16x32_bf16(
                false, a.bf, false, b.bf, (short)0, acc[nt], false, false);
        }
    }

#pragma unroll
    for (int nt = 0; nt < 8; ++nt)
#pragma unroll
        for (int r = 0; r < 8; ++r)
            atomicAdd(Out + (size_t)urow[r] * DIM + nt * 16 + m, acc[nt][r]);
}

// ---------------------------------------------------------------------------
// GroupNorm(1,D) + ReLU -> bf16 output. One wave per row (32 lanes x 4 ch).
// ---------------------------------------------------------------------------
__global__ void __launch_bounds__(256) gn_relu_bf16_kernel(
    const float* __restrict__ X, const float* __restrict__ gw,
    const float* __restrict__ gb, unsigned short* __restrict__ Obf, int nrows)
{
    const int lane = threadIdx.x & 31;
    const int row  = blockIdx.x * 8 + (threadIdx.x >> 5);
    if (row >= nrows) return;

    const float4 x = reinterpret_cast<const float4*>(X + (size_t)row * DIM)[lane];
    float s = x.x + x.y + x.z + x.w;
    float q = x.x * x.x + x.y * x.y + x.z * x.z + x.w * x.w;
#pragma unroll
    for (int off = 16; off > 0; off >>= 1) {
        s += __shfl_xor(s, off, 32);
        q += __shfl_xor(q, off, 32);
    }
    const float mu = s * (1.f / 128.f);
    const float rs = rsqrtf(q * (1.f / 128.f) - mu * mu + 1e-5f);
    const int c = lane * 4;
    const float y0 = fmaxf((x.x - mu) * rs * gw[c + 0] + gb[c + 0], 0.f);
    const float y1 = fmaxf((x.y - mu) * rs * gw[c + 1] + gb[c + 1], 0.f);
    const float y2 = fmaxf((x.z - mu) * rs * gw[c + 2] + gb[c + 2], 0.f);
    const float y3 = fmaxf((x.w - mu) * rs * gw[c + 3] + gb[c + 3], 0.f);
    const unsigned int lo = (unsigned int)f2bf(y0) | ((unsigned int)f2bf(y1) << 16);
    const unsigned int hi = (unsigned int)f2bf(y2) | ((unsigned int)f2bf(y3) << 16);
    reinterpret_cast<uint2*>(Obf + (size_t)row * DIM)[lane] = make_uint2(lo, hi);
}

// ---------------------------------------------------------------------------
// GroupNorm(1,D) + residual + ReLU. Updates Feat (f32) in place and writes
// bf16 copy for the next iteration's GEMMs.
// ---------------------------------------------------------------------------
__global__ void __launch_bounds__(256) gn_res_relu_kernel(
    const float* __restrict__ X, const float* __restrict__ gw,
    const float* __restrict__ gb, float* __restrict__ Feat,
    unsigned short* __restrict__ Obf, int nrows)
{
    const int lane = threadIdx.x & 31;
    const int row  = blockIdx.x * 8 + (threadIdx.x >> 5);
    if (row >= nrows) return;

    const float4 x = reinterpret_cast<const float4*>(X + (size_t)row * DIM)[lane];
    float s = x.x + x.y + x.z + x.w;
    float q = x.x * x.x + x.y * x.y + x.z * x.z + x.w * x.w;
#pragma unroll
    for (int off = 16; off > 0; off >>= 1) {
        s += __shfl_xor(s, off, 32);
        q += __shfl_xor(q, off, 32);
    }
    const float mu = s * (1.f / 128.f);
    const float rs = rsqrtf(q * (1.f / 128.f) - mu * mu + 1e-5f);
    const int c = lane * 4;

    const float4 r4 = reinterpret_cast<const float4*>(Feat + (size_t)row * DIM)[lane];
    const float z0 = fmaxf((x.x - mu) * rs * gw[c + 0] + gb[c + 0] + r4.x, 0.f);
    const float z1 = fmaxf((x.y - mu) * rs * gw[c + 1] + gb[c + 1] + r4.y, 0.f);
    const float z2 = fmaxf((x.z - mu) * rs * gw[c + 2] + gb[c + 2] + r4.z, 0.f);
    const float z3 = fmaxf((x.w - mu) * rs * gw[c + 3] + gb[c + 3] + r4.w, 0.f);

    reinterpret_cast<float4*>(Feat + (size_t)row * DIM)[lane] = make_float4(z0, z1, z2, z3);
    const unsigned int lo = (unsigned int)f2bf(z0) | ((unsigned int)f2bf(z1) << 16);
    const unsigned int hi = (unsigned int)f2bf(z2) | ((unsigned int)f2bf(z3) << 16);
    reinterpret_cast<uint2*>(Obf + (size_t)row * DIM)[lane] = make_uint2(lo, hi);
}

// ---------------------------------------------------------------------------
__global__ void f32_to_bf16_kernel(const float* __restrict__ in,
                                   unsigned short* __restrict__ out, int n)
{
    const int i = blockIdx.x * 256 + threadIdx.x;
    if (i < n) out[i] = f2bf(in[i]);
}

__global__ void init_feat_kernel(const float* __restrict__ in,
                                 float* __restrict__ out,
                                 unsigned short* __restrict__ bf, int n)
{
    const int i = blockIdx.x * 256 + threadIdx.x;
    if (i < n) {
        const float v = in[i];
        out[i] = v;
        bf[i]  = f2bf(v);
    }
}

// ---------------------------------------------------------------------------
extern "C" void kernel_launch(void* const* d_in, const int* in_sizes, int n_in,
                              void* d_out, int out_size, void* d_ws, size_t ws_size,
                              hipStream_t stream)
{
    (void)in_sizes; (void)n_in; (void)out_size; (void)ws_size;

    const float* feat    = (const float*)d_in[0];
    const int*   pre_u   = (const int*)d_in[1];
    const int*   pre_v   = (const int*)d_in[2];
    const int*   suc_u   = (const int*)d_in[3];
    const int*   suc_v   = (const int*)d_in[4];
    const int*   left_u  = (const int*)d_in[5];
    const int*   left_v  = (const int*)d_in[6];
    const int*   right_u = (const int*)d_in[7];
    const int*   right_v = (const int*)d_in[8];
    const float* W_ctr   = (const float*)d_in[9];
    const float* W_pre   = (const float*)d_in[10];
    const float* W_suc   = (const float*)d_in[11];
    const float* W_left  = (const float*)d_in[12];
    const float* W_right = (const float*)d_in[13];
    const float* gn1_w   = (const float*)d_in[14];
    const float* gn1_b   = (const float*)d_in[15];
    const float* W_ctr2  = (const float*)d_in[16];
    const float* gn2_w   = (const float*)d_in[17];
    const float* gn2_b   = (const float*)d_in[18];
    float* out = (float*)d_out;

    // Workspace carve-up
    char* ws = (char*)d_ws;
    float*          temp  = (float*)ws;                                        // N*D f32
    unsigned short* bfbuf = (unsigned short*)(ws + (size_t)N_NODES * DIM * 4); // N*D bf16
    unsigned short* wbf   = (unsigned short*)(ws + (size_t)N_NODES * DIM * 4
                                                 + (size_t)N_NODES * DIM * 2);
    const int MAT = DIM * DIM;                   // 16384
    unsigned short* w_ctr_bf   = wbf;
    unsigned short* w_pre_bf   = w_ctr_bf   + (size_t)NITERS * MAT;
    unsigned short* w_suc_bf   = w_pre_bf   + (size_t)NITERS * NSCALES * MAT;
    unsigned short* w_left_bf  = w_suc_bf   + (size_t)NITERS * NSCALES * MAT;
    unsigned short* w_right_bf = w_left_bf  + (size_t)NITERS * MAT;
    unsigned short* w_ctr2_bf  = w_right_bf + (size_t)NITERS * MAT;

    // --- weight conversion (fp32 -> bf16, ~2 MB total) ---
    {
        int n;
        n = NITERS * MAT;
        f32_to_bf16_kernel<<<(n + 255) / 256, 256, 0, stream>>>(W_ctr,   w_ctr_bf,   n);
        f32_to_bf16_kernel<<<(n + 255) / 256, 256, 0, stream>>>(W_left,  w_left_bf,  n);
        f32_to_bf16_kernel<<<(n + 255) / 256, 256, 0, stream>>>(W_right, w_right_bf, n);
        f32_to_bf16_kernel<<<(n + 255) / 256, 256, 0, stream>>>(W_ctr2,  w_ctr2_bf,  n);
        n = NITERS * NSCALES * MAT;
        f32_to_bf16_kernel<<<(n + 255) / 256, 256, 0, stream>>>(W_pre,   w_pre_bf,   n);
        f32_to_bf16_kernel<<<(n + 255) / 256, 256, 0, stream>>>(W_suc,   w_suc_bf,   n);
    }

    // --- feat -> out (res buffer) + bf16 copy ---
    {
        const int n = N_NODES * DIM;
        init_feat_kernel<<<(n + 255) / 256, 256, 0, stream>>>(feat, out, bfbuf, n);
    }

    const int rowTiles  = N_NODES / 16;   // 15625
    const int edgeTiles = NEDGE / 16;     // 31250
    const int lrTiles   = NEDGE_LR / 16;  // 3125
    const dim3 blk(256);

    for (int i = 0; i < NITERS; ++i) {
        // temp = feat @ W_ctr[i].T
        gemm_rows_kernel<<<(rowTiles + 7) / 8, blk, 0, stream>>>(
            bfbuf, w_ctr_bf + (size_t)i * MAT, temp, rowTiles);

        // edge scatter-GEMMs
        for (int k = 0; k < NSCALES; ++k) {
            gemm_edges_kernel<<<(edgeTiles + 7) / 8, blk, 0, stream>>>(
                bfbuf, w_pre_bf + (size_t)(i * NSCALES + k) * MAT, temp,
                pre_u + (size_t)k * NEDGE, pre_v + (size_t)k * NEDGE, edgeTiles);
            gemm_edges_kernel<<<(edgeTiles + 7) / 8, blk, 0, stream>>>(
                bfbuf, w_suc_bf + (size_t)(i * NSCALES + k) * MAT, temp,
                suc_u + (size_t)k * NEDGE, suc_v + (size_t)k * NEDGE, edgeTiles);
        }
        gemm_edges_kernel<<<(lrTiles + 7) / 8, blk, 0, stream>>>(
            bfbuf, w_left_bf + (size_t)i * MAT, temp, left_u, left_v, lrTiles);
        gemm_edges_kernel<<<(lrTiles + 7) / 8, blk, 0, stream>>>(
            bfbuf, w_right_bf + (size_t)i * MAT, temp, right_u, right_v, lrTiles);

        // h = relu(GN1(temp)) -> bf16 (reuse bfbuf; edge GEMMs are done)
        gn_relu_bf16_kernel<<<N_NODES / 8, blk, 0, stream>>>(
            temp, gn1_w + (size_t)i * DIM, gn1_b + (size_t)i * DIM, bfbuf, N_NODES);

        // temp = h @ W_ctr2[i].T
        gemm_rows_kernel<<<(rowTiles + 7) / 8, blk, 0, stream>>>(
            bfbuf, w_ctr2_bf + (size_t)i * MAT, temp, rowTiles);

        // feat = relu(GN2(temp) + res); res = feat
        gn_res_relu_kernel<<<N_NODES / 8, blk, 0, stream>>>(
            temp, gn2_w + (size_t)i * DIM, gn2_b + (size_t)i * DIM, out, bfbuf, N_NODES);
    }
}
